// MixtureOfExperts_55044300866197
// MI455X (gfx1250) — matrix operand order repaired and checked by reading, not compiled, not run on hardware
//
#include <hip/hip_runtime.h>
#include <hip/hip_bf16.h>
#include <math.h>

// ---------------- problem constants (fixed by the reference) ----------------
#define N_TOK 8192      // B*T = 4*2048
#define C_DIM 1024
#define E_NUM 8
#define F_DIM 4096
#define MT    32        // tokens per workgroup tile
#define FK    64        // F-chunk per fused up/down step
#define KT1   128       // W1 K-tile staged per round (4 x 32)

typedef __attribute__((ext_vector_type(16))) __bf16 v16bf;
typedef __attribute__((ext_vector_type(8)))  float  v8f;

// fp32 -> bf16 via native hardware conversion (RNE)
__device__ __forceinline__ unsigned short bf16bits(float a) {
    return __builtin_bit_cast(unsigned short, (__bf16)a);
}
__device__ __forceinline__ unsigned int pack2bf(float a, float b) {
    return (unsigned int)bf16bits(a) | ((unsigned int)bf16bits(b) << 16);
}

__device__ __forceinline__ float gelu_exact(float x) {
    return 0.5f * x * (1.0f + erff(x * 0.70710678118654752440f));
}

// A-matrix per-lane element order within a 32-wide K chunk (matches B):
// lane-half 0 streams k = {0..15}, half 1 streams k = {16..31}.
// pos32A maps k-in-chunk -> storage position; consecutive for k % 4 == 0 groups.
__device__ __forceinline__ int pos32A(int j) {
    return j;
}

// A fragment (16x32 bf16): rows are pre-swizzled per 32-chunk, so each lane
// reads 16 contiguous bf16 (32 B) -> 2x ds_load_b128.
template <int RSTRIDE>
__device__ __forceinline__ v16bf load_a_sw(const unsigned short* base,
                                           int row0, int k0, int lane) {
    const int m    = row0 + (lane & 15);
    const int half = (lane >> 4) & 1;
    return *(const v16bf*)(base + m * RSTRIDE + k0 + half * 16);
}

// B fragment (32x16 bf16): tile stored column-major (pos = n*32 + k), so each
// lane reads 16 contiguous bf16 (32 B) -> 2x ds_load_b128.
__device__ __forceinline__ v16bf load_b_cm(const unsigned short* chunk,
                                           int n0, int lane) {
    const int n    = n0 + (lane & 15);
    const int half = (lane >> 4) & 1;
    return *(const v16bf*)(chunk + n * 32 + half * 16);
}

__device__ __forceinline__ v8f wmma_bf16(v16bf a, v16bf b, v8f c) {
    return __builtin_amdgcn_wmma_f32_16x16x32_bf16(
        /*neg_a=*/false, a, /*neg_b=*/false, b,
        /*c_mod=*/(short)0, c, /*reuse_a=*/false, /*reuse_b=*/false);
}

// ---------------------------------------------------------------------------
// Kernel 0: zero output (atomically accumulated) and per-expert counters.
// ---------------------------------------------------------------------------
__global__ void zero_kernel(float* __restrict__ out, int n, int* __restrict__ cnt) {
    for (int i = blockIdx.x * blockDim.x + threadIdx.x; i < n;
         i += gridDim.x * blockDim.x)
        out[i] = 0.0f;
    if (blockIdx.x == 0 && threadIdx.x < E_NUM) cnt[threadIdx.x] = 0;
}

// ---------------------------------------------------------------------------
// Kernel 1: router. One thread per token: logits -> softmax -> top-2 ->
// renormalize -> scatter (token, weight) into per-expert lists.
// ---------------------------------------------------------------------------
__global__ __launch_bounds__(128) void router_kernel(
    const float* __restrict__ x, const float* __restrict__ gw,
    int* __restrict__ cnt, int* __restrict__ perm, float* __restrict__ cwl) {
    const int t = blockIdx.x * 128 + threadIdx.x;
    if (t >= N_TOK) return;

    float lg[E_NUM];
#pragma unroll
    for (int e = 0; e < E_NUM; ++e) lg[e] = 0.0f;

    const float* xr = x + (size_t)t * C_DIM;
    for (int c = 0; c < C_DIM; c += 4) {
        const float4 xv = *(const float4*)(xr + c);
#pragma unroll
        for (int e = 0; e < E_NUM; ++e) {
            lg[e] += xv.x * gw[(c + 0) * E_NUM + e];
            lg[e] += xv.y * gw[(c + 1) * E_NUM + e];
            lg[e] += xv.z * gw[(c + 2) * E_NUM + e];
            lg[e] += xv.w * gw[(c + 3) * E_NUM + e];
        }
    }

    float mx = lg[0];
#pragma unroll
    for (int e = 1; e < E_NUM; ++e) mx = fmaxf(mx, lg[e]);
    float p[E_NUM];
#pragma unroll
    for (int e = 0; e < E_NUM; ++e) p[e] = expf(lg[e] - mx);

    int i0 = 0;
#pragma unroll
    for (int e = 1; e < E_NUM; ++e) if (p[e] > p[i0]) i0 = e;   // lowest idx on tie
    int i1 = (i0 == 0) ? 1 : 0;
#pragma unroll
    for (int e = 0; e < E_NUM; ++e)
        if (e != i0 && p[e] > p[i1]) i1 = e;

    const float inv = 1.0f / (p[i0] + p[i1]);
    const float w0 = p[i0] * inv, w1 = p[i1] * inv;

    int s0 = atomicAdd(&cnt[i0], 1);
    perm[i0 * N_TOK + s0] = t;
    cwl[i0 * N_TOK + s0] = w0;
    int s1 = atomicAdd(&cnt[i1], 1);
    perm[i1 * N_TOK + s1] = t;
    cwl[i1 * N_TOK + s1] = w1;
}

// ---------------------------------------------------------------------------
// Kernel 2: fused expert MLP. block = (tile of 32 tokens) x (expert).
// out[tok,:] += cw * gelu(x[tok,:] @ W1_e) @ W2_e, F tiled by FK=64.
// 8 wave32s: each wave owns output cols [w*128, w*128+128) = 2x8 16x16 tiles.
// ---------------------------------------------------------------------------
__global__ __launch_bounds__(256) void moe_expert_kernel(
    const float* __restrict__ x, const float* __restrict__ w1,
    const float* __restrict__ w2, const int* __restrict__ cnt,
    const int* __restrict__ perm, const float* __restrict__ cwl,
    float* __restrict__ out) {
    const int e     = blockIdx.y;
    const int count = cnt[e];
    const int mbase = blockIdx.x * MT;
    if (mbase >= count) return;

    extern __shared__ unsigned char smem[];
    unsigned short* Xs  = (unsigned short*)smem;   // [32][1024] A-swizzled tokens
    unsigned short* W2s = Xs + MT * C_DIM;         // [1024][32] col-major W2 k-tile
    unsigned short* W1s = W2s + 32 * C_DIM;        // 4 x [64][32] col-major W1 tiles
    unsigned short* Hs  = W1s + KT1 * FK;          // [32][64] A-swizzled gelu(h)
    int*   toks = (int*)(Hs + MT * FK);            // [32]
    float* cws  = (float*)(toks + MT);             // [32]

    const int tid  = threadIdx.x;
    const int lane = tid & 31;
    const int w    = tid >> 5;   // wave 0..7

    if (tid < MT) {
        const int idx = mbase + tid;
        const bool ok = idx < count;
        toks[tid] = ok ? perm[e * N_TOK + idx] : -1;
        cws[tid]  = ok ? cwl[e * N_TOK + idx] : 0.0f;
    }
    __syncthreads();

    // gather token rows, convert, store A-swizzled (float4 -> ds_store_b64)
#pragma unroll 4
    for (int i = 0; i < (MT * C_DIM) / (256 * 4); ++i) {
        const int idx4 = i * 256 + tid;
        const int m = idx4 >> 8;            // 256 float4 per row
        const int c = (idx4 & 255) * 4;
        const int tok = toks[m];
        float4 xv;
        if (tok >= 0) xv = *(const float4*)(x + (size_t)tok * C_DIM + c);
        else          xv = make_float4(0.f, 0.f, 0.f, 0.f);
        const int pos = m * C_DIM + (c & ~31) + pos32A(c & 31);
        *(uint2*)(Xs + pos) = make_uint2(pack2bf(xv.x, xv.y), pack2bf(xv.z, xv.w));
    }
    __syncthreads();

    v8f oacc[2][8];
#pragma unroll
    for (int a = 0; a < 2; ++a)
#pragma unroll
        for (int b = 0; b < 8; ++b) oacc[a][b] = {};

    const int hmt = w >> 2;   // 16-row M-tile of h this wave produces
    const int hnt = w & 3;    // 16-col N-tile of h
    const size_t w1_base = (size_t)e * C_DIM * F_DIM;
    const size_t w2_base = (size_t)e * F_DIM * C_DIM;

    for (int fc = 0; fc < F_DIM / FK; ++fc) {
        // ---- up-projection: h[32x64] = X[32x1024] @ W1[:, fc*64 .. +64)
        v8f hacc = {};
        for (int kt = 0; kt < C_DIM / KT1; ++kt) {   // 8 staging rounds
            __syncthreads();   // previous W1s consumers done
            {
                // thread -> (column n, k-chunk kc); 32 coalesced k-strided loads,
                // pack into 4x ds_store_b128 (column-major chunk layout)
                const int n  = tid & 63;
                const int kc = tid >> 6;             // 0..3
                const float* g = w1 + w1_base +
                                 (size_t)(kt * KT1 + kc * 32) * F_DIM + fc * FK + n;
                unsigned int pk[16];
#pragma unroll
                for (int kk = 0; kk < 32; kk += 2)
                    pk[kk >> 1] = pack2bf(g[(size_t)kk * F_DIM],
                                          g[(size_t)(kk + 1) * F_DIM]);
                uint4* dst = (uint4*)(W1s + kc * (32 * FK) + n * 32);
#pragma unroll
                for (int q = 0; q < 4; ++q)
                    dst[q] = make_uint4(pk[4*q], pk[4*q+1], pk[4*q+2], pk[4*q+3]);
                if (kt + 1 < C_DIM / KT1)
                    __builtin_prefetch(g + (size_t)KT1 * F_DIM, 0, 1);
            }
            __syncthreads();
#pragma unroll
            for (int kc = 0; kc < 4; ++kc) {
                const v16bf a = load_a_sw<C_DIM>(Xs, hmt * 16, kt * KT1 + kc * 32, lane);
                const v16bf b = load_b_cm(W1s + kc * (32 * FK), hnt * 16, lane);
                hacc = wmma_bf16(a, b, hacc);
            }
        }
        __syncthreads();

        // exact GELU, convert, stage h tile A-swizzled (C/D layout: m = half*8+v)
#pragma unroll
        for (int v = 0; v < 8; ++v) {
            const int ml = ((lane >> 4) & 1) * 8 + v;
            const int m  = hmt * 16 + ml;
            const int k  = hnt * 16 + (lane & 15);
            Hs[m * FK + (k & ~31) + pos32A(k & 31)] = bf16bits(gelu_exact(hacc[v]));
        }
        __syncthreads();

        // ---- down-projection: oacc[32x1024] += Hs[32x64] @ W2[fc*64 .. , :]
        for (int kt2 = 0; kt2 < FK / 32; ++kt2) {
            {
                // thread -> 4 columns; 32 coalesced k-strided loads per column,
                // pack into 4x ds_store_b128 (column-major layout)
                const float* g = w2 + w2_base +
                                 (size_t)(fc * FK + kt2 * 32) * C_DIM;
#pragma unroll
                for (int q = 0; q < 4; ++q) {
                    const int n = tid + q * 256;
                    unsigned int pk[16];
#pragma unroll
                    for (int kk = 0; kk < 32; kk += 2)
                        pk[kk >> 1] = pack2bf(g[(size_t)kk * C_DIM + n],
                                              g[(size_t)(kk + 1) * C_DIM + n]);
                    uint4* dst = (uint4*)(W2s + n * 32);
#pragma unroll
                    for (int r = 0; r < 4; ++r)
                        dst[r] = make_uint4(pk[4*r], pk[4*r+1], pk[4*r+2], pk[4*r+3]);
                }
            }
            __syncthreads();
            const v16bf a0 = load_a_sw<FK>(Hs, 0,  kt2 * 32, lane);
            const v16bf a1 = load_a_sw<FK>(Hs, 16, kt2 * 32, lane);
#pragma unroll
            for (int j = 0; j < 8; ++j) {
                const v16bf b = load_b_cm(W2s, w * 128 + j * 16, lane);
                oacc[0][j] = wmma_bf16(a0, b, oacc[0][j]);
                oacc[1][j] = wmma_bf16(a1, b, oacc[1][j]);
            }
            __syncthreads();   // W2s safe to overwrite
        }
    }

    // weighted scatter-add into out (exactly 2 contributions per element ->
    // float atomics are order-independent, hence deterministic)
#pragma unroll
    for (int mt = 0; mt < 2; ++mt) {
#pragma unroll
        for (int j = 0; j < 8; ++j) {
#pragma unroll
            for (int v = 0; v < 8; ++v) {
                const int ml = mt * 16 + ((lane >> 4) & 1) * 8 + v;
                const int tok = toks[ml];
                if (tok >= 0) {
                    const int c = w * 128 + j * 16 + (lane & 15);
                    atomicAdd(&out[(size_t)tok * C_DIM + c], cws[ml] * oacc[mt][j][v]);
                }
            }
        }
    }
}

// ---------------------------------------------------------------------------
extern "C" void kernel_launch(void* const* d_in, const int* in_sizes, int n_in,
                              void* d_out, int out_size, void* d_ws, size_t ws_size,
                              hipStream_t stream) {
    const float* x  = (const float*)d_in[0];   // [4,2048,1024]
    const float* gw = (const float*)d_in[1];   // [1024,8]
    const float* w1 = (const float*)d_in[2];   // [8,1024,4096]
    const float* w2 = (const float*)d_in[3];   // [8,4096,1024]
    float* out = (float*)d_out;                // [4,2048,1024]

    // workspace: counters + per-expert token lists + weights (~512 KB)
    int*   cnt  = (int*)d_ws;
    int*   perm = cnt + 16;                    // [E][N]
    float* cwl  = (float*)(perm + E_NUM * N_TOK);

    zero_kernel<<<512, 256, 0, stream>>>(out, N_TOK * C_DIM, cnt);
    router_kernel<<<N_TOK / 128, 128, 0, stream>>>(x, gw, cnt, perm, cwl);

    const size_t smem =
        (size_t)(MT * C_DIM + 32 * C_DIM + KT1 * FK + MT * FK) * sizeof(unsigned short)
        + MT * sizeof(int) + MT * sizeof(float);   // ~148.5 KB (2 WGs fit per WGP)
    dim3 grid(N_TOK / MT, E_NUM);
    moe_expert_kernel<<<grid, 256, smem, stream>>>(x, w1, w2, cnt, perm, cwl, out);
}
